// BiSelfAttention_56315611185786
// MI455X (gfx1250) — matrix operand order repaired and checked by reading, not compiled, not run on hardware
//
#include <hip/hip_runtime.h>
#include <hip/hip_bf16.h>
#include <cstdint>
#include <cstddef>

#define D_MODEL 1024
#define N_HEADS 16
#define D_HEAD  64
#define BATCH   4
#define SEQ     2048
#define MROWS   (BATCH * SEQ)   // 8192

typedef __bf16 bf16_t;
typedef bf16_t v16bf __attribute__((ext_vector_type(16)));
typedef bf16_t v8bf  __attribute__((ext_vector_type(8)));
typedef float  v8f   __attribute__((ext_vector_type(8)));
typedef unsigned int u32x4 __attribute__((ext_vector_type(4)));
typedef int i32x4 __attribute__((ext_vector_type(4)));
typedef int i32x8 __attribute__((ext_vector_type(8)));

static __device__ __forceinline__ v16bf concat8(v8bf lo, v8bf hi) {
  return __builtin_shufflevector(lo, hi, 0,1,2,3,4,5,6,7,8,9,10,11,12,13,14,15);
}

// ---------------------------------------------------------------------------
// Tensor Data Mover: 2-D tile load (global -> LDS), bf16 elements.
// D# packing per CDNA5 ISA (group0) / §8.4 (group1).
// clang-23 toolchain builtin arity: (u32x4, i32x8, i32x4, i32x4, i32x8, i32).
// ---------------------------------------------------------------------------
static __device__ __forceinline__ void tdm_load_tile_2d(
    unsigned lds_off, const void* gptr,
    unsigned tile_d0, unsigned tile_d1,
    unsigned tensor_d0, unsigned tensor_d1,
    unsigned stride0_elems) {
  unsigned long long ga = (unsigned long long)(uintptr_t)gptr;
  u32x4 g0;
  g0[0] = 1u;                                                   // count=1
  g0[1] = lds_off;                                              // lds_addr
  g0[2] = (unsigned)(ga & 0xffffffffull);                       // gaddr lo
  g0[3] = ((unsigned)((ga >> 32) & 0x1ffffffull)) | 0x80000000u;// gaddr hi | type=2
  i32x8 g1;
  g1[0] = (int)(1u << 16);                                      // data_size = 2B
  g1[1] = (int)((tensor_d0 & 0xffffu) << 16);                   // dim0[15:0] @ b63:48
  g1[2] = (int)((tensor_d0 >> 16) | ((tensor_d1 & 0xffffu) << 16));
  g1[3] = (int)((tensor_d1 >> 16) | (tile_d0 << 16));           // tile_dim0 @ b127:112
  g1[4] = (int)(tile_d1 & 0xffffu);                             // tile_dim1, tile_dim2=0
  g1[5] = (int)stride0_elems;                                   // dim0_stride[31:0]
  g1[6] = 0;
  g1[7] = 0;
  i32x4 z4; z4[0] = 0; z4[1] = 0; z4[2] = 0; z4[3] = 0;
  i32x8 z8;
#pragma unroll
  for (int i = 0; i < 8; ++i) z8[i] = 0;
  __builtin_amdgcn_tensor_load_to_lds(g0, g1, z4, z4, z8, 0);
}

// ---------------------------------------------------------------------------
// f32 -> bf16 cast
// ---------------------------------------------------------------------------
__global__ void cast_f32_bf16(const float* __restrict__ in,
                              bf16_t* __restrict__ out, int n) {
  int i = blockIdx.x * blockDim.x + threadIdx.x;
  if (i < n) out[i] = (bf16_t)in[i];
}

// ---------------------------------------------------------------------------
// Tiled bf16 GEMM: C[M,N] = A[M,K] * Bw[N,K]^T  (Bw row-major [N,K])
// 256 threads (8 waves), 128x128 tile; wave = 64x32 (4x2 C tiles).
// Double-buffered TDM staging: DMA of tile k+1 overlaps WMMAs on tile k.
// EPI==0: QKV scatter epilogue (q/k [BH,T,Dh] bf16, v transposed [BH,Dh,T]).
// EPI==1: plain f32 store [M,N].
// ---------------------------------------------------------------------------
template<int EPI>
__global__ void __launch_bounds__(256)
gemm_bf16(const bf16_t* __restrict__ A, const bf16_t* __restrict__ Bw,
          int M, int N, int K,
          bf16_t* __restrict__ qo, bf16_t* __restrict__ ko,
          bf16_t* __restrict__ vto, float* __restrict__ fo) {
  __shared__ __align__(16) bf16_t At[2][128][32];
  __shared__ __align__(16) bf16_t Bt[2][128][32];

  const int tid  = threadIdx.x;
  const int lane = tid & 31;
  const int w    = tid >> 5;
  const int wr   = w >> 2;          // 0..1 : m offset wr*64
  const int wc   = w & 3;           // 0..3 : n offset wc*32
  const int bm   = blockIdx.y;
  const int bn   = blockIdx.x;
  const int hi   = lane >> 4;
  const int ln   = lane & 15;
  const int kbA  = hi * 16;         // A fragment: 16 contiguous K at kbA
  const int kbB  = hi * 16;         // B fragment: 16 contiguous K at kbB

  v8f acc[4][2] = {};

  const bf16_t* Abase = A  + (size_t)bm * 128 * K;
  const bf16_t* Bbase = Bw + (size_t)bn * 128 * K;
  const int KT = K >> 5;

  if (w == 0) {
    tdm_load_tile_2d((unsigned)(size_t)&At[0][0][0], Abase,
                     32u, 128u, (unsigned)K, 128u, (unsigned)K);
    tdm_load_tile_2d((unsigned)(size_t)&Bt[0][0][0], Bbase,
                     32u, 128u, (unsigned)K, 128u, (unsigned)K);
  }

  for (int kt = 0; kt < KT; ++kt) {
    const int cur = kt & 1;
    if (w == 0) __builtin_amdgcn_s_wait_tensorcnt(0);  // tile kt resident
    __syncthreads();  // tile kt visible; buf[1-cur] free (tile kt-1 consumed)
    if (w == 0 && kt + 1 < KT) {
      const int k0n = (kt + 1) << 5;
      tdm_load_tile_2d((unsigned)(size_t)&At[1 - cur][0][0], Abase + k0n,
                       32u, 128u, (unsigned)(K - k0n), 128u, (unsigned)K);
      tdm_load_tile_2d((unsigned)(size_t)&Bt[1 - cur][0][0], Bbase + k0n,
                       32u, 128u, (unsigned)(K - k0n), 128u, (unsigned)K);
    }

    v16bf am[4];
#pragma unroll
    for (int i = 0; i < 4; ++i) {
      int row = wr * 64 + i * 16 + ln;
      am[i] = concat8(*(const v8bf*)&At[cur][row][kbA],
                      *(const v8bf*)&At[cur][row][kbA + 8]);
    }
    v16bf bb[2];
#pragma unroll
    for (int j = 0; j < 2; ++j) {
      int nr = wc * 32 + j * 16 + ln;
      bb[j] = concat8(*(const v8bf*)&Bt[cur][nr][kbB],
                      *(const v8bf*)&Bt[cur][nr][kbB + 8]);
    }
#pragma unroll
    for (int i = 0; i < 4; ++i)
#pragma unroll
      for (int j = 0; j < 2; ++j)
        acc[i][j] = __builtin_amdgcn_wmma_f32_16x16x32_bf16(
            false, am[i], false, bb[j], (short)0, acc[i][j], false, false);
  }

  // Epilogue. C layout: lane ln holds column; VGPR r holds row r + 8*hi.
#pragma unroll
  for (int i = 0; i < 4; ++i) {
#pragma unroll
    for (int j = 0; j < 2; ++j) {
      int n     = bn * 128 + wc * 32 + j * 16 + ln;
      int mbase = bm * 128 + wr * 64 + i * 16 + 8 * hi;
#pragma unroll
      for (int r = 0; r < 8; ++r) {
        int   m   = mbase + r;
        float val = acc[i][j][r];
        if (EPI == 0) {
          int sel = n >> 10;
          int rem = n & 1023;
          int h   = rem >> 6;
          int dh  = rem & 63;
          int b   = m >> 11;
          int t   = m & 2047;
          int bh  = b * N_HEADS + h;
          if (sel == 0)      qo[((size_t)bh * SEQ + t) * D_HEAD + dh]  = (bf16_t)val;
          else if (sel == 1) ko[((size_t)bh * SEQ + t) * D_HEAD + dh]  = (bf16_t)val;
          else               vto[((size_t)bh * D_HEAD + dh) * SEQ + t] = (bf16_t)val;
        } else {
          fo[(size_t)m * N + n] = val;
        }
      }
    }
  }
}

// ---------------------------------------------------------------------------
// Flash attention, fully transposed: grid (B*H, T/64), block 128 (4 waves).
// Wave owns 16 queries (= lanes); computes S^T = K*Q^T so per-query softmax
// stats are per-lane scalars (one shfl_xor(16) per reduction), and
// O^T = V^T * P^T so V^T fragments are contiguous loads. No LDS needed;
// P^T C-layout -> B-fragment needs only 8 cross-half shuffles.
// ---------------------------------------------------------------------------
__global__ void __launch_bounds__(128)
attn_fwd(const bf16_t* __restrict__ Q, const bf16_t* __restrict__ Kb,
         const bf16_t* __restrict__ Vt, bf16_t* __restrict__ Y) {
  const int tid  = threadIdx.x;
  const int lane = tid & 31;
  const int w    = tid >> 5;
  const int hi   = lane >> 4;
  const int ln   = lane & 15;
  const int bh   = blockIdx.x;
  const int b    = bh / N_HEADS;
  const int h    = bh % N_HEADS;
  const int q0   = blockIdx.y * 64 + w * 16;
  const int kbA  = hi * 16;         // A fragment contiguous K
  const int kbB  = hi * 16;         // B fragment contiguous K

  const bf16_t* qp = Q  + (size_t)bh * SEQ * D_HEAD;
  const bf16_t* kp = Kb + (size_t)bh * SEQ * D_HEAD;
  const bf16_t* vp = Vt + (size_t)bh * D_HEAD * SEQ;

  // Q as B-fragments (col = query ln, K-dim = dh), two tiles cover Dh=64.
  v16bf bq[2];
#pragma unroll
  for (int f = 0; f < 2; ++f) {
    const bf16_t* base = qp + (size_t)(q0 + ln) * D_HEAD + f * 32 + kbB;
    bq[f] = concat8(*(const v8bf*)base, *(const v8bf*)(base + 8));
  }

  float mi = -1.0e30f, li = 0.0f;
  v8f o[4] = {};   // O^T: rows = dh (4 blocks of 16), cols = queries

  for (int j0 = 0; j0 < SEQ; j0 += 32) {
    // S^T tiles: rows = keys (jb*16 block), cols = queries.
    v8f s[2] = {};
#pragma unroll
    for (int jb = 0; jb < 2; ++jb) {
#pragma unroll
      for (int f = 0; f < 2; ++f) {
        const bf16_t* base =
            kp + (size_t)(j0 + jb * 16 + ln) * D_HEAD + f * 32 + kbA;
        v16bf ak = concat8(*(const v8bf*)base, *(const v8bf*)(base + 8));
        s[jb] = __builtin_amdgcn_wmma_f32_16x16x32_bf16(
            false, ak, false, bq[f], (short)0, s[jb], false, false);
      }
      s[jb] = s[jb] * 0.125f;   // 1/sqrt(Dh)
    }

    // Per-lane (= per-query) online softmax.
    // Lane holds keys jb*16 + 8*hi + r for r=0..7, tiles jb=0,1.
    float mloc = fmaxf(s[0][0], s[1][0]);
#pragma unroll
    for (int r = 1; r < 8; ++r)
      mloc = fmaxf(mloc, fmaxf(s[0][r], s[1][r]));
    mloc = fmaxf(mloc, __shfl_xor(mloc, 16, 32));   // other key half
    float mn    = fmaxf(mi, mloc);
    float alpha = __expf(mi - mn);
    mi = mn;

    float p0[8], p1[8];
    float sloc = 0.0f;
#pragma unroll
    for (int r = 0; r < 8; ++r) {
      p0[r] = __expf(s[0][r] - mn);
      p1[r] = __expf(s[1][r] - mn);
      sloc += p0[r] + p1[r];
    }
    sloc += __shfl_xor(sloc, 16, 32);
    li = li * alpha + sloc;

    // Rescale running output (scalar broadcast per lane).
#pragma unroll
    for (int d = 0; d < 4; ++d) o[d] = o[d] * alpha;

    // P^T as B-fragment: lane needs 16 consecutive keys (hi*16 + j) of its
    // query column. half0 owns keys 0-7 (p0), needs 8-15 from half1's p0;
    // half1 owns keys 24-31 (p1), needs 16-23 from half0's p1.
    v16bf bp;
#pragma unroll
    for (int r = 0; r < 8; ++r) {
      float sendv = hi ? p0[r] : p1[r];
      float got   = __shfl_xor(sendv, 16, 32);
      float e0 = hi ? got  : p0[r];   // j = r
      float e1 = hi ? p1[r] : got;    // j = 8 + r
      bp[r]     = (bf16_t)e0;
      bp[8 + r] = (bf16_t)e1;
    }

    // O^T += V^T * P^T : V^T rows are dh, contiguous along keys (A frag).
#pragma unroll
    for (int d = 0; d < 4; ++d) {
      const bf16_t* base = vp + (size_t)(d * 16 + ln) * SEQ + j0 + kbA;
      v16bf av = concat8(*(const v8bf*)base, *(const v8bf*)(base + 8));
      o[d] = __builtin_amdgcn_wmma_f32_16x16x32_bf16(
          false, av, false, bp, (short)0, o[d], false, false);
    }
  }

  // Normalize, store y[b, t=q0+ln, h*64 + d*16 + r + 8*hi] as bf16.
  float inv = 1.0f / li;
  const int t = q0 + ln;
  bf16_t* yrow = Y + ((size_t)b * SEQ + t) * D_MODEL + h * 64;
#pragma unroll
  for (int d = 0; d < 4; ++d)
#pragma unroll
    for (int r = 0; r < 8; ++r)
      yrow[d * 16 + r + 8 * hi] = (bf16_t)(o[d][r] * inv);
}

// ---------------------------------------------------------------------------
// Workspace layout (bytes):
//   xb   @ 0         : 8192*1024*2  = 16777216
//   wqb  @ 16777216  : 3072*1024*2  =  6291456
//   wpb  @ 23068672  : 1024*1024*2  =  2097152
//   qb   @ 25165824  : 64*2048*64*2 = 16777216
//   kb   @ 41943040  : 16777216
//   vtb  @ 58720256  : 16777216
//   yb   @ 75497472  : 16777216     (end 92274688 ~ 88 MiB)
// ---------------------------------------------------------------------------
extern "C" void kernel_launch(void* const* d_in, const int* in_sizes, int n_in,
                              void* d_out, int out_size, void* d_ws, size_t ws_size,
                              hipStream_t stream) {
  const float* x     = (const float*)d_in[0];
  const float* Wqkv  = (const float*)d_in[1];
  const float* Wproj = (const float*)d_in[2];
  float*       out   = (float*)d_out;
  char*        ws    = (char*)d_ws;

  bf16_t* xb  = (bf16_t*)(ws + 0);
  bf16_t* wqb = (bf16_t*)(ws + 16777216);
  bf16_t* wpb = (bf16_t*)(ws + 23068672);
  bf16_t* qb  = (bf16_t*)(ws + 25165824);
  bf16_t* kb  = (bf16_t*)(ws + 41943040);
  bf16_t* vtb = (bf16_t*)(ws + 58720256);
  bf16_t* yb  = (bf16_t*)(ws + 75497472);

  cast_f32_bf16<<<dim3((8388608 + 255) / 256), 256, 0, stream>>>(x, xb, 8388608);
  cast_f32_bf16<<<dim3((3145728 + 255) / 256), 256, 0, stream>>>(Wqkv, wqb, 3145728);
  cast_f32_bf16<<<dim3((1048576 + 255) / 256), 256, 0, stream>>>(Wproj, wpb, 1048576);

  // qkv = x @ Wqkv^T, scattered into q/k (row-major) and v (transposed)
  gemm_bf16<0><<<dim3(3072 / 128, 8192 / 128), 256, 0, stream>>>(
      xb, wqb, MROWS, 3 * D_MODEL, D_MODEL, qb, kb, vtb, nullptr);

  // flash attention per (b,h), 64 query rows per workgroup
  attn_fwd<<<dim3(BATCH * N_HEADS, SEQ / 64), 128, 0, stream>>>(qb, kb, vtb, yb);

  // out = y @ Wproj^T, f32
  gemm_bf16<1><<<dim3(1024 / 128, 8192 / 128), 256, 0, stream>>>(
      yb, wpb, MROWS, D_MODEL, D_MODEL, nullptr, nullptr, nullptr, out);
}
